// RWKVTimeMixing_10900626997665
// MI455X (gfx1250) — matrix operand order repaired and checked by reading, not compiled, not run on hardware
//
#include <hip/hip_runtime.h>
#include <hip/hip_bf16.h>

// ---------------------------------------------------------------------------
// RWKV time-mixing for MI455X (gfx1250, wave32, WMMA).
//   * GEMMs: v_wmma_f32_16x16x32_bf16, 32x64 tile per wave, B staged in LDS
//     (async global->LDS, double buffered, ASYNCcnt-tracked).
//   * WKV scan: 3-pass chunked associative scan (16x parallelism).
// ---------------------------------------------------------------------------

#define Bb 4
#define Ss 2048
#define Dd 1024
#define Mrows (Bb * Ss)            // 8192 GEMM rows

#define BLK_M 256                  // block output rows (8 waves x 32)
#define BLK_N 64                   // block output cols
#define KSTEP 32                   // bf16 WMMA K per step
#define NKIT  (Dd / KSTEP)         // 32 K iterations

#define CHUNKS 16                  // WKV scan chunks
#define CLEN   (Ss / CHUNKS)       // 128 steps per chunk

typedef __attribute__((ext_vector_type(16))) __bf16 v16bf;
typedef __attribute__((ext_vector_type(8)))  float  v8f;

union Frag16 {                     // 16 bf16 = 32 bytes = two b128 loads
    uint4 q[2];
    v16bf bf;
};

__device__ __forceinline__ unsigned short f2bf(float f) {
    unsigned u = __builtin_bit_cast(unsigned, f);
    u += 0x7FFFu + ((u >> 16) & 1u);          // round-to-nearest-even
    return (unsigned short)(u >> 16);
}
__device__ __forceinline__ float bf2f(unsigned short h) {
    return __builtin_bit_cast(float, ((unsigned)h) << 16);
}

// ---- async global->LDS copy (gfx1250 path), with safe fallback -------------
#if defined(__has_builtin)
#if __has_builtin(__builtin_amdgcn_global_load_async_to_lds_b128)
#define USE_ASYNC_LDS 1
#endif
#endif

typedef int v4i_vs __attribute__((vector_size(4 * sizeof(int))));
typedef __attribute__((address_space(1))) v4i_vs* gv4i_p;   // global int4*
typedef __attribute__((address_space(3))) v4i_vs* lv4i_p;   // LDS int4*

__device__ __forceinline__ void stage16(const unsigned short* g,
                                        unsigned short* l) {
#if defined(USE_ASYNC_LDS)
    // global_load_async_to_lds_b128 : tracked by ASYNCcnt
    __builtin_amdgcn_global_load_async_to_lds_b128((gv4i_p)g, (lv4i_p)l, 0, 0);
#else
    *(uint4*)l = *(const uint4*)g;            // load + ds_store fallback
#endif
}

__device__ __forceinline__ void wait_stage() {
#if defined(USE_ASYNC_LDS)
#if __has_builtin(__builtin_amdgcn_s_wait_asynccnt)
    __builtin_amdgcn_s_wait_asynccnt(0);
#else
    asm volatile("s_wait_asynccnt 0x0" ::: "memory");
#endif
#endif
}

// ---- kernel 1: weight f32 -> bf16 -----------------------------------------
__global__ void cvt_w_kernel(const float* __restrict__ src,
                             unsigned short* __restrict__ dst, int n) {
    int i = blockIdx.x * blockDim.x + threadIdx.x;
    if (i < n) dst[i] = f2bf(src[i]);
}

// ---- kernel 2: token shift + time mix -> xk/xv/xr (bf16) ------------------
__global__ void mix_shift_kernel(const float* __restrict__ x,
                                 const float* __restrict__ tmk,
                                 const float* __restrict__ tmv,
                                 const float* __restrict__ tmr,
                                 unsigned short* __restrict__ xk,
                                 unsigned short* __restrict__ xv,
                                 unsigned short* __restrict__ xr) {
    int i = blockIdx.x * blockDim.x + threadIdx.x;   // over B*S*D
    if (i >= Bb * Ss * Dd) return;
    int d = i & (Dd - 1);
    int s = (i >> 10) & (Ss - 1);
    float xc = x[i];
    float sh = (s == 0) ? 0.0f : x[i - Dd];
    float mk = tmk[d], mv = tmv[d], mr = tmr[d];
    xk[i] = f2bf(xc * mk + sh * (1.0f - mk));
    xv[i] = f2bf(xc * mv + sh * (1.0f - mv));
    xr[i] = f2bf(xc * mr + sh * (1.0f - mr));
}

// ---- WMMA GEMM  y[m,e] = sum_d A[m,d] * W[e,d] ----------------------------
// MODE 0: store f32.   MODE 1: store bf16(sigmoid(y)).
// Block 256 threads = 8 waves; block tile 256x64; wave tile 32x64.
// B tile (64 cols x 32 K) double-buffered in LDS via async copy.
template <int MODE>
__global__ __launch_bounds__(256)
void gemm_bf16_kernel(const unsigned short* __restrict__ A,   // [M][D] bf16
                      const unsigned short* __restrict__ W,   // [E][D] bf16
                      float* __restrict__ outF,               // MODE 0
                      unsigned short* __restrict__ outB) {    // MODE 1
    __shared__ __align__(16) unsigned short Bs[2][BLK_N * KSTEP];  // 2 x 4KB

    const int lane = threadIdx.x & 31;
    const int wave = threadIdx.x >> 5;
    const int half = lane >> 4;                    // 0 | 1 (half-wave)
    const int l16  = lane & 15;
    const int m0 = blockIdx.y * BLK_M + wave * 32;
    const int n0 = blockIdx.x * BLK_N;

    // Cooperative B staging: thread t copies 16 bytes (8 bf16) per K-step.
    const int scol = threadIdx.x >> 2;             // 0..63
    const int sk   = (threadIdx.x & 3) * 8;        // 0,8,16,24
    const unsigned short* bSrc = W + (size_t)(n0 + scol) * Dd + sk;
    unsigned short* bDst0 = &Bs[0][scol * KSTEP + sk];
    unsigned short* bDst1 = &Bs[1][scol * KSTEP + sk];

    // A fragments (16x32 bf16, ISA 7.12.2): lane row = l16,
    //   half 0 -> K base +0, half 1 -> K base +8; second group at +16.
    const unsigned short* aRow0 = A + (size_t)(m0 + l16) * Dd + half * 8;
    const unsigned short* aRow1 = aRow0 + (size_t)16 * Dd;

    v8f c[2][4];
#pragma unroll
    for (int mt = 0; mt < 2; ++mt)
#pragma unroll
        for (int t = 0; t < 4; ++t) c[mt][t] = {};

    stage16(bSrc, bDst0);                          // prologue: stage K-step 0

    for (int it = 0; it < NKIT; ++it) {
        const int kk = it * KSTEP;
        wait_stage();                              // own async copy landed
        __syncthreads();                           // tile visible; prev reads done
        if (it + 1 < NKIT)
            stage16(bSrc + (kk + KSTEP), (it & 1) ? bDst0 : bDst1);
        if (it + 2 < NKIT) {                       // prefetch A stream
            __builtin_prefetch(aRow0 + kk + 2 * KSTEP, 0, 0);
            __builtin_prefetch(aRow1 + kk + 2 * KSTEP, 0, 0);
        }

        Frag16 a0, a1;
        a0.q[0] = *(const uint4*)(aRow0 + kk);
        a0.q[1] = *(const uint4*)(aRow0 + kk + 16);
        a1.q[0] = *(const uint4*)(aRow1 + kk);
        a1.q[1] = *(const uint4*)(aRow1 + kk + 16);

        const unsigned short* bs = &Bs[it & 1][0];
#pragma unroll
        for (int t = 0; t < 4; ++t) {
            // B fragment (32x16): lane col = l16, half 0 -> K +0, half 1 -> K +8; second group at +16.
            Frag16 b;
            const unsigned short* bp = bs + (t * 16 + l16) * KSTEP + half * 8;
            b.q[0] = *(const uint4*)(bp);
            b.q[1] = *(const uint4*)(bp + 16);
            c[0][t] = __builtin_amdgcn_wmma_f32_16x16x32_bf16(
                false, a0.bf, false, b.bf, (short)0, c[0][t], false, false);
            c[1][t] = __builtin_amdgcn_wmma_f32_16x16x32_bf16(
                false, a1.bf, false, b.bf, (short)0, c[1][t], false, false);
        }
    }

    // C/D layout: VGPR r -> M = r + 8*half, N = l16.
#pragma unroll
    for (int mt = 0; mt < 2; ++mt) {
#pragma unroll
        for (int t = 0; t < 4; ++t) {
#pragma unroll
            for (int r = 0; r < 8; ++r) {
                const int row = m0 + mt * 16 + r + 8 * half;
                const int col = n0 + t * 16 + l16;
                float val = c[mt][t][r];
                if (MODE == 0) {
                    outF[(size_t)row * Dd + col] = val;
                } else {
                    float sg = 1.0f / (1.0f + __expf(-val));
                    outB[(size_t)row * Dd + col] = f2bf(sg);
                }
            }
        }
    }
}

// ---- WKV chunked scan ------------------------------------------------------
// Recurrence (per b,d):  A <- e^w A + e^k v ;  B <- e^w B + e^k, held in
// stabilized (a,b,p) form:  A = a*e^p.  Over a chunk of length L:
//   A_out = e^{wL} A_in + A_local  => associative; scan over 16 chunks.

// pass 1: per-chunk local accumulation from zero state (parallel B*D*CHUNKS)
__global__ __launch_bounds__(256)
void wkv_part1_kernel(const float* __restrict__ k, const float* __restrict__ v,
                      const float* __restrict__ td,
                      float* __restrict__ ca, float* __restrict__ cb,
                      float* __restrict__ cp) {
    int gid = blockIdx.x * blockDim.x + threadIdx.x;
    if (gid >= Bb * Dd * CHUNKS) return;
    const int d = gid & (Dd - 1);
    const int j = (gid >> 10) & (CHUNKS - 1);
    const int b = gid >> 14;
    const float w = -__expf(td[d]);

    float aa = 0.0f, bb = 0.0f, pp = -1e30f;
    size_t idx = ((size_t)b * Ss + (size_t)j * CLEN) * Dd + d;
    for (int s = 0; s < CLEN; ++s, idx += Dd) {
        const float kk = k[idx];
        const float vv = v[idx];
        float ww2 = w + pp;
        float p2  = fmaxf(ww2, kk);
        float e1  = __expf(ww2 - p2);
        float e2  = __expf(kk - p2);
        aa = e1 * aa + e2 * vv;
        bb = e1 * bb + e2;
        pp = p2;
    }
    size_t o = ((size_t)b * CHUNKS + j) * Dd + d;
    ca[o] = aa; cb[o] = bb; cp[o] = pp;
}

// pass 2: serial exclusive prefix over the 16 chunk summaries (parallel B*D)
__global__ __launch_bounds__(256)
void wkv_part2_kernel(const float* __restrict__ td,
                      float* __restrict__ ca, float* __restrict__ cb,
                      float* __restrict__ cp) {
    int gid = blockIdx.x * blockDim.x + threadIdx.x;
    if (gid >= Bb * Dd) return;
    const int d = gid & (Dd - 1);
    const int b = gid >> 10;
    const float wL = -__expf(td[d]) * (float)CLEN;

    float aa = 0.0f, bb = 0.0f, pp = -1e30f;
    for (int j = 0; j < CHUNKS; ++j) {
        size_t o = ((size_t)b * CHUNKS + j) * Dd + d;
        float la = ca[o], lb = cb[o], lp = cp[o];
        ca[o] = aa; cb[o] = bb; cp[o] = pp;       // exclusive prefix state
        float pw = pp + wL;
        float pn = fmaxf(pw, lp);
        float e1 = __expf(pw - pn);
        float e2 = __expf(lp - pn);
        aa = e1 * aa + e2 * la;
        bb = e1 * bb + e2 * lb;
        pp = pn;
    }
}

// pass 3: replay each chunk from its prefix state, emit rwkv (parallel B*D*C)
__global__ __launch_bounds__(256)
void wkv_part3_kernel(const float* __restrict__ k, const float* __restrict__ v,
                      const unsigned short* __restrict__ rbf,
                      const float* __restrict__ td, const float* __restrict__ tf,
                      const float* __restrict__ ca, const float* __restrict__ cb,
                      const float* __restrict__ cp,
                      unsigned short* __restrict__ rwkv) {
    int gid = blockIdx.x * blockDim.x + threadIdx.x;
    if (gid >= Bb * Dd * CHUNKS) return;
    const int d = gid & (Dd - 1);
    const int j = (gid >> 10) & (CHUNKS - 1);
    const int b = gid >> 14;
    const float w = -__expf(td[d]);
    const float u = tf[d];

    size_t o = ((size_t)b * CHUNKS + j) * Dd + d;
    float aa = ca[o], bb = cb[o], pp = cp[o];

    size_t idx = ((size_t)b * Ss + (size_t)j * CLEN) * Dd + d;
    for (int s = 0; s < CLEN; ++s, idx += Dd) {
        const float kk = k[idx];
        const float vv = v[idx];
        float ww = u + kk;
        float p  = fmaxf(pp, ww);
        float e1 = __expf(pp - p);
        float e2 = __expf(ww - p);
        float out = (e1 * aa + e2 * vv) / (e1 * bb + e2);
        rwkv[idx] = f2bf(bf2f(rbf[idx]) * out);
        float ww2 = w + pp;
        float p2  = fmaxf(ww2, kk);
        float e1b = __expf(ww2 - p2);
        float e2b = __expf(kk - p2);
        aa = e1b * aa + e2b * vv;
        bb = e1b * bb + e2b;
        pp = p2;
    }
}

// ---------------------------------------------------------------------------
extern "C" void kernel_launch(void* const* d_in, const int* in_sizes, int n_in,
                              void* d_out, int out_size, void* d_ws, size_t ws_size,
                              hipStream_t stream) {
    const float* x   = (const float*)d_in[0];
    const float* tmk = (const float*)d_in[1];
    const float* tmv = (const float*)d_in[2];
    const float* tmr = (const float*)d_in[3];
    const float* td  = (const float*)d_in[4];
    const float* tf  = (const float*)d_in[5];
    const float* Wk  = (const float*)d_in[6];
    const float* Wv  = (const float*)d_in[7];
    const float* Wr  = (const float*)d_in[8];
    const float* Wo  = (const float*)d_in[9];

    const size_t MB = 1024ull * 1024ull;
    unsigned char* base = (unsigned char*)d_ws;
    // bf16 weights: 2 MB each
    unsigned short* Wkb = (unsigned short*)(base + 0 * MB);
    unsigned short* Wvb = (unsigned short*)(base + 2 * MB);
    unsigned short* Wrb = (unsigned short*)(base + 4 * MB);
    unsigned short* Wob = (unsigned short*)(base + 6 * MB);
    // bf16 activations: 16 MB each
    unsigned short* xk  = (unsigned short*)(base + 8 * MB);
    unsigned short* xv  = (unsigned short*)(base + 24 * MB);
    unsigned short* xr  = (unsigned short*)(base + 40 * MB);
    // f32 k,v: 32 MB each; bf16 sigmoid(r): 16 MB
    float*          kf  = (float*)(base + 56 * MB);
    float*          vf  = (float*)(base + 88 * MB);
    unsigned short* rb  = (unsigned short*)(base + 120 * MB);
    // chunk summaries (a,b,p): 256 KB each
    float*          ca  = (float*)(base + 136 * MB);
    float*          cb  = (float*)(base + 137 * MB);
    float*          cp  = (float*)(base + 138 * MB);
    // rwkv (bf16) aliases xk, which is dead after the k-GEMM.
    unsigned short* rwkv = xk;

    // 1) weights -> bf16
    {
        int n = Dd * Dd;
        dim3 g((n + 255) / 256), blk(256);
        cvt_w_kernel<<<g, blk, 0, stream>>>(Wk, Wkb, n);
        cvt_w_kernel<<<g, blk, 0, stream>>>(Wv, Wvb, n);
        cvt_w_kernel<<<g, blk, 0, stream>>>(Wr, Wrb, n);
        cvt_w_kernel<<<g, blk, 0, stream>>>(Wo, Wob, n);
    }
    // 2) token shift + mix
    {
        int n = Bb * Ss * Dd;
        mix_shift_kernel<<<(n + 255) / 256, 256, 0, stream>>>(
            x, tmk, tmv, tmr, xk, xv, xr);
    }
    // 3) three projection GEMMs
    {
        dim3 grid(Dd / BLK_N, Mrows / BLK_M), blk(256);
        gemm_bf16_kernel<0><<<grid, blk, 0, stream>>>(xk, Wkb, kf, nullptr);
        gemm_bf16_kernel<0><<<grid, blk, 0, stream>>>(xv, Wvb, vf, nullptr);
        gemm_bf16_kernel<1><<<grid, blk, 0, stream>>>(xr, Wrb, nullptr, rb);
    }
    // 4) WKV chunked scan (+ fused sigmoid(r)*wkv in pass 3)
    {
        int n1 = Bb * Dd * CHUNKS;
        wkv_part1_kernel<<<(n1 + 255) / 256, 256, 0, stream>>>(
            kf, vf, td, ca, cb, cp);
        int n2 = Bb * Dd;
        wkv_part2_kernel<<<(n2 + 255) / 256, 256, 0, stream>>>(td, ca, cb, cp);
        wkv_part3_kernel<<<(n1 + 255) / 256, 256, 0, stream>>>(
            kf, vf, rb, td, tf, ca, cb, cp, rwkv);
    }
    // 5) output GEMM -> d_out (f32)
    {
        dim3 grid(Dd / BLK_N, Mrows / BLK_M), blk(256);
        gemm_bf16_kernel<0><<<grid, blk, 0, stream>>>(
            rwkv, Wob, (float*)d_out, nullptr);
    }
}
